// NAMAttention_55138790146722
// MI455X (gfx1250) — compile-verified, matrix operand order checked by reading, not run on hardware
//
#include <hip/hip_runtime.h>
#include <hip/hip_bf16.h>
#include <cstdint>

typedef __attribute__((ext_vector_type(16))) _Float16 v16h;
typedef __attribute__((ext_vector_type(4)))  _Float16 v4h;
typedef __attribute__((ext_vector_type(8)))  float    v8f;

#define S_LEN 2048
#define B_SZ  8
#define DM    1024
#define NH    16
#define HD    64
#define MTOK  (S_LEN * B_SZ)          // 16384 tokens

// ---------------------------------------------------------------------------
// CDNA5 async copy: global -> LDS, 128 bits per lane, tracked by ASYNCcnt.
// VDST operand = 32-bit LDS byte address; generic LDS pointers carry the LDS
// offset in their low 32 bits (ISA 10.2 aperture rules), so truncation works.
// ---------------------------------------------------------------------------
__device__ __forceinline__ void async_ld_b128(const void* gptr, const void* lptr) {
    unsigned lds_off = (unsigned)(uintptr_t)lptr;
    asm volatile("global_load_async_to_lds_b128 %0, %1, off"
                 :: "v"(lds_off), "v"(gptr) : "memory");
}
#if __has_builtin(__builtin_amdgcn_s_wait_asynccnt)
#define WAIT_ASYNC(n) __builtin_amdgcn_s_wait_asynccnt(n)
#else
#define WAIT_ASYNC(n) asm volatile("s_wait_asynccnt %0" :: "n"(n) : "memory")
#endif

// ---------------------------------------------------------------------------
// Fragment loader matching CDNA5 16-bit A/B WMMA layouts (ISA 7.12.2):
// lane 0-15 hold row (lane&15); lanes 16-31 hold the same rows with K+8.
// Halves 0..7 = K k0..k0+7, halves 8..15 = K k0+16..k0+23, k0 = (lane>=16)*8.
// ---------------------------------------------------------------------------
__device__ __forceinline__ v16h load_frag(const _Float16* tile, int row0, int ld) {
    const int lane = threadIdx.x & 31;
    const _Float16* p = tile + (size_t)(row0 + (lane & 15)) * ld + ((lane >> 4) << 3);
    v16h r;
    ((float4*)&r)[0] = *(const float4*)(p);
    ((float4*)&r)[1] = *(const float4*)(p + 16);
    return r;
}

__device__ __forceinline__ v8f vzero8() { v8f z = {0.f,0.f,0.f,0.f,0.f,0.f,0.f,0.f}; return z; }

// ---------------------------------------------------------------------------
// fp32 -> fp16 conversion, float4 vectorized (used for input and weights)
// ---------------------------------------------------------------------------
__global__ __launch_bounds__(256) void cvt_f32_f16(const float* __restrict__ X,
                                                   _Float16* __restrict__ Xh) {
    size_t i = (size_t)blockIdx.x * 256 + threadIdx.x;
    float4 v = ((const float4*)X)[i];
    v4h h; h.x = (_Float16)v.x; h.y = (_Float16)v.y; h.z = (_Float16)v.z; h.w = (_Float16)v.w;
    ((v4h*)Xh)[i] = h;
}

// ---------------------------------------------------------------------------
// GEMM: Y[M,N] = X_h[M,K] @ W_h[N,K]^T + bias, f32 accumulate.
// Tile 128x128x32, 256 threads = 8 waves; wave = 64(M) x 32(N) = 4x2 WMMA tiles.
// Triple-buffered LDS fed by GLOBAL_LOAD_ASYNC_TO_LDS_B128 (ASYNCcnt pipeline):
//   iter kt: issue batch kt+1 -> buf (kt+1)%3 ; s_wait_asynccnt 4 (batch kt
//   retired, in-order); barrier; 8x WMMA from buf kt%3.
// Buffer (kt+1)%3 was last read at iter kt-2 -> one barrier (iter kt-1) in between.
// ---------------------------------------------------------------------------
template<bool OUT_F16>
__global__ __launch_bounds__(256)
void gemm_xwT(const _Float16* __restrict__ X, const _Float16* __restrict__ Wh,
              const float* __restrict__ bias, _Float16* __restrict__ Yh,
              float* __restrict__ Yf, int M, int N, int K) {
    constexpr int BM = 128, BN = 128, BK = 32;
    __shared__ _Float16 sX[3][BM * BK];
    __shared__ _Float16 sW[3][BN * BK];

    const int t    = threadIdx.x;
    const int lane = t & 31;
    const int wv   = t >> 5;                  // 0..7
    const int mBlk = blockIdx.x * BM;
    const int nBlk = blockIdx.y * BN;
    const int wm   = (wv >> 2) * 64;          // 0 / 64
    const int wn   = (wv & 3) * 32;           // 0 / 32 / 64 / 96

    v8f acc[4][2];
    #pragma unroll
    for (int i = 0; i < 4; ++i)
        #pragma unroll
        for (int j = 0; j < 2; ++j) acc[i][j] = vzero8();

    const int nk = K / BK;

    // 4 async b128 copies per thread per stage (2 X chunks + 2 W chunks)
    auto issue = [&](int kt, int buf) {
        #pragma unroll
        for (int i = 0; i < 2; ++i) {
            int c = t * 2 + i;                 // 0..511
            int row = c >> 2, kc = (c & 3) * 8;
            async_ld_b128(X + (size_t)(mBlk + row) * K + kt * BK + kc,
                          &sX[buf][row * BK + kc]);
            async_ld_b128(Wh + (size_t)(nBlk + row) * K + kt * BK + kc,
                          &sW[buf][row * BK + kc]);
        }
    };

    issue(0, 0);
    for (int kt = 0; kt < nk; ++kt) {
        if (kt + 1 < nk) { issue(kt + 1, (kt + 1) % 3); WAIT_ASYNC(4); }
        else             { WAIT_ASYNC(0); }
        __syncthreads();
        const int b = kt % 3;
        v16h aF[4], bF[2];
        #pragma unroll
        for (int i = 0; i < 4; ++i) aF[i] = load_frag(sX[b], wm + i * 16, BK);
        #pragma unroll
        for (int j = 0; j < 2; ++j) bF[j] = load_frag(sW[b], wn + j * 16, BK);
        #pragma unroll
        for (int i = 0; i < 4; ++i)
            #pragma unroll
            for (int j = 0; j < 2; ++j)
                acc[i][j] = __builtin_amdgcn_wmma_f32_16x16x32_f16(
                    false, aF[i], false, bF[j], (short)0, acc[i][j], false, false);
    }

    // Epilogue: C layout -> element (M = vg + 8*(lane>=16), N = lane&15)
    float bv[2];
    #pragma unroll
    for (int j = 0; j < 2; ++j) bv[j] = bias[nBlk + wn + j * 16 + (lane & 15)];
    const int mLane = mBlk + wm + ((lane >> 4) << 3);
    const int nLane = nBlk + wn + (lane & 15);
    #pragma unroll
    for (int i = 0; i < 4; ++i)
        #pragma unroll
        for (int j = 0; j < 2; ++j)
            #pragma unroll
            for (int vg = 0; vg < 8; ++vg) {
                int m = mLane + i * 16 + vg;
                int nn = nLane + j * 16;
                float val = acc[i][j][vg] + bv[j];
                if (OUT_F16) Yh[(size_t)m * N + nn] = (_Float16)val;
                else         Yf[(size_t)m * N + nn] = val;
            }
}

// ---------------------------------------------------------------------------
// w = sigmoid(X@Ww^T + bw), r = sigmoid(X@Wr^T + br): one wave per dot (fp32)
// ---------------------------------------------------------------------------
__global__ __launch_bounds__(256)
void wr_proj(const float* __restrict__ X,
             const float* __restrict__ Ww, const float* __restrict__ Wb,
             const float* __restrict__ Rw, const float* __restrict__ Rb,
             float* __restrict__ wout, float* __restrict__ rout) {
    int g    = blockIdx.x * 8 + (threadIdx.x >> 5);   // 2^19 waves
    int lane = threadIdx.x & 31;
    int tsel = g >> 18;
    int idx  = g & ((1 << 18) - 1);
    int m = idx >> 4, h = idx & 15;
    const float* Wp = (tsel ? Rw : Ww) + (size_t)h * DM;
    const float* xp = X + (size_t)m * DM;
    float s = 0.f;
    for (int i = lane; i < DM; i += 32) s += xp[i] * Wp[i];
    #pragma unroll
    for (int o = 16; o > 0; o >>= 1) s += __shfl_xor(s, o, 32);
    if (lane == 0) {
        float b = tsel ? Rb[h] : Wb[h];
        float y = 1.f / (1.f + __expf(-(s + b)));
        (tsel ? rout : wout)[m * NH + h] = y;
    }
}

// ---------------------------------------------------------------------------
// Unit-normalize Q and K per head (64-wide) in place; wave per (tensor,m,head)
// ---------------------------------------------------------------------------
__global__ __launch_bounds__(256)
void norm_qk(_Float16* __restrict__ Qh, _Float16* __restrict__ Kh) {
    int g    = blockIdx.x * 8 + (threadIdx.x >> 5);
    int lane = threadIdx.x & 31;
    int tsel = g >> 18;
    int idx  = g & ((1 << 18) - 1);
    int m = idx >> 4, h = idx & 15;
    _Float16* p = (tsel ? Kh : Qh) + (size_t)m * DM + h * HD;
    float a = (float)p[lane], c = (float)p[lane + 32];
    float s = a * a + c * c;
    #pragma unroll
    for (int o = 16; o > 0; o >>= 1) s += __shfl_xor(s, o, 32);
    float inv = 1.f / fmaxf(sqrtf(s), 1e-12f);
    p[lane]      = (_Float16)(a * inv);
    p[lane + 32] = (_Float16)(c * inv);
}

// ---------------------------------------------------------------------------
// Vh *= w (per token, per head)
// ---------------------------------------------------------------------------
__global__ __launch_bounds__(256)
void scale_v(_Float16* __restrict__ Vh, const float* __restrict__ wbuf) {
    size_t i = (size_t)blockIdx.x * 256 + threadIdx.x;   // MTOK*DM
    int m = (int)(i >> 10);
    int h = (int)((i & 1023) >> 6);
    Vh[i] = (_Float16)((float)Vh[i] * wbuf[m * NH + h]);
}

// ---------------------------------------------------------------------------
// A[b,n,v,q] = sum_s Vw[s,(b,n),v] * K[s,(b,n),q]  — one block per (b,n),
// WMMA with K-dim = s in chunks of 32, LDS tiles stored transposed [feat][s].
// Writes f32 A to d_out region and f16 copy to workspace.
// ---------------------------------------------------------------------------
__global__ __launch_bounds__(256)
void accum_A(const _Float16* __restrict__ Kh, const _Float16* __restrict__ Vh,
             float* __restrict__ Af, _Float16* __restrict__ Ah) {
    const int bn = blockIdx.x;          // b*16 + n
    const int b  = bn >> 4, n = bn & 15;
    __shared__ _Float16 sK[2][HD * 32]; // [q][s]
    __shared__ _Float16 sV[2][HD * 32]; // [v][s]

    const int t = threadIdx.x, lane = t & 31, wv = t >> 5;   // 8 waves
    const int vt   = wv >> 1;           // v-tile 0..3
    const int qoff = (wv & 1) * 32;     // q half 0/32

    v8f acc[2] = {vzero8(), vzero8()};
    uint4 kreg, vreg;
    const int s_i = t >> 3, f8 = (t & 7) * 8;
    const int nsc = S_LEN / 32;

    auto fetchA = [&](int sc) {
        size_t m = (size_t)(sc * 32 + s_i) * B_SZ + b;
        kreg = *(const uint4*)(Kh + m * DM + n * HD + f8);
        vreg = *(const uint4*)(Vh + m * DM + n * HD + f8);
    };
    auto stageA = [&](int bb) {
        union { uint4 u; _Float16 h[8]; } ck, cv;
        ck.u = kreg; cv.u = vreg;
        #pragma unroll
        for (int e = 0; e < 8; ++e) {
            sK[bb][(f8 + e) * 32 + s_i] = ck.h[e];   // transpose at store
            sV[bb][(f8 + e) * 32 + s_i] = cv.h[e];
        }
    };

    fetchA(0); stageA(0);
    for (int sc = 0; sc < nsc; ++sc) {
        __syncthreads();
        if (sc + 1 < nsc) fetchA(sc + 1);
        const int bb = sc & 1;
        v16h aF = load_frag(sV[bb], vt * 16, 32);
        v16h bF0 = load_frag(sK[bb], qoff, 32);
        v16h bF1 = load_frag(sK[bb], qoff + 16, 32);
        acc[0] = __builtin_amdgcn_wmma_f32_16x16x32_f16(false, aF, false, bF0, (short)0, acc[0], false, false);
        acc[1] = __builtin_amdgcn_wmma_f32_16x16x32_f16(false, aF, false, bF1, (short)0, acc[1], false, false);
        if (sc + 1 < nsc) stageA((sc + 1) & 1);
    }

    #pragma unroll
    for (int j = 0; j < 2; ++j)
        #pragma unroll
        for (int vg = 0; vg < 8; ++vg) {
            int vIdx = vt * 16 + vg + ((lane >> 4) << 3);
            int qIdx = qoff + j * 16 + (lane & 15);
            size_t idx = ((size_t)bn * HD + vIdx) * HD + qIdx;
            float val = acc[j][vg];
            Af[idx] = val;
            Ah[idx] = (_Float16)val;
        }
}

// ---------------------------------------------------------------------------
// Mid[s,(b,n),v] = r * sum_q Q[s,(b,n),q] * A[b,n,v,q] — WMMA, fragments
// loaded straight from global (Q rows strided by B*DM, A rows contiguous).
// ---------------------------------------------------------------------------
__global__ __launch_bounds__(256)
void apply_A(const _Float16* __restrict__ Qh, const _Float16* __restrict__ Ah,
             const float* __restrict__ rbuf, _Float16* __restrict__ Mid) {
    const int bn = blockIdx.x;          // b*16+n
    const int b  = bn >> 4, n = bn & 15;
    const int lane  = threadIdx.x & 31;
    const int sBase = blockIdx.y * 128 + (threadIdx.x >> 5) * 16;   // 8 waves x 16 rows

    const _Float16* Abase = Ah + (size_t)bn * (HD * HD);
    v8f acc[4] = {vzero8(), vzero8(), vzero8(), vzero8()};

    #pragma unroll
    for (int kt = 0; kt < 2; ++kt) {    // K = 64 = 2 x 32
        const _Float16* qbase = Qh + ((size_t)sBase * B_SZ + b) * DM + n * HD + kt * 32;
        v16h aF = load_frag(qbase, 0, B_SZ * DM);        // row step = next s token
        #pragma unroll
        for (int j = 0; j < 4; ++j) {
            v16h bF = load_frag(Abase + kt * 32, j * 16, HD);
            acc[j] = __builtin_amdgcn_wmma_f32_16x16x32_f16(
                false, aF, false, bF, (short)0, acc[j], false, false);
        }
    }

    float rv[8];
    #pragma unroll
    for (int vg = 0; vg < 8; ++vg) {
        int s = sBase + vg + ((lane >> 4) << 3);
        rv[vg] = rbuf[((size_t)s * B_SZ + b) * NH + n];
    }
    #pragma unroll
    for (int j = 0; j < 4; ++j)
        #pragma unroll
        for (int vg = 0; vg < 8; ++vg) {
            int s = sBase + vg + ((lane >> 4) << 3);
            size_t m = (size_t)s * B_SZ + b;
            Mid[m * DM + n * HD + j * 16 + (lane & 15)] = (_Float16)(acc[j][vg] * rv[vg]);
        }
}

// ---------------------------------------------------------------------------
// Launch
// ---------------------------------------------------------------------------
extern "C" void kernel_launch(void* const* d_in, const int* in_sizes, int n_in,
                              void* d_out, int out_size, void* d_ws, size_t ws_size,
                              hipStream_t stream) {
    (void)in_sizes; (void)n_in; (void)out_size; (void)ws_size;
    const float* inp  = (const float*)d_in[0];
    const float* Wq_w = (const float*)d_in[1];  const float* Wq_b = (const float*)d_in[2];
    const float* Wk_w = (const float*)d_in[3];  const float* Wk_b = (const float*)d_in[4];
    const float* Wv_w = (const float*)d_in[5];  const float* Wv_b = (const float*)d_in[6];
    const float* Wo_w = (const float*)d_in[7];  const float* Wo_b = (const float*)d_in[8];
    const float* Ww_w = (const float*)d_in[9];  const float* Ww_b = (const float*)d_in[10];
    const float* Wr_w = (const float*)d_in[11]; const float* Wr_b = (const float*)d_in[12];

    float* out_f = (float*)d_out;                       // [S,B,DM]
    float* A_f   = out_f + (size_t)MTOK * DM;           // [B,NH,64,64]

    char* ws = (char*)d_ws;
    constexpr size_t HBYTES = (size_t)MTOK * DM * sizeof(_Float16);   // 32 MiB
    constexpr size_t GATE   = (size_t)MTOK * NH * 4;                  // 1 MiB
    constexpr size_t WBYTES = (size_t)DM * DM * sizeof(_Float16);     // 2 MiB
    _Float16* Xh  = (_Float16*)(ws);
    _Float16* Qh  = (_Float16*)(ws + HBYTES);
    _Float16* Kh  = (_Float16*)(ws + 2 * HBYTES);
    _Float16* Vh  = (_Float16*)(ws + 3 * HBYTES);
    float*    wbf = (float*)   (ws + 4 * HBYTES);
    float*    rbf = (float*)   (ws + 4 * HBYTES + GATE);
    _Float16* Ah  = (_Float16*)(ws + 4 * HBYTES + 2 * GATE);
    _Float16* Mid = (_Float16*)(ws + 4 * HBYTES + 2 * GATE + (size_t)B_SZ * NH * HD * HD * 2);
    _Float16* Wqh = (_Float16*)(ws + 5 * HBYTES + 3 * GATE);
    _Float16* Wkh = (_Float16*)(ws + 5 * HBYTES + 3 * GATE + WBYTES);
    _Float16* Wvh = (_Float16*)(ws + 5 * HBYTES + 3 * GATE + 2 * WBYTES);
    _Float16* Woh = (_Float16*)(ws + 5 * HBYTES + 3 * GATE + 3 * WBYTES);

    // 1) fp32 -> fp16: input + four big weight matrices
    cvt_f32_f16<<<(MTOK * DM / 4) / 256, 256, 0, stream>>>(inp, Xh);
    cvt_f32_f16<<<(DM * DM / 4) / 256, 256, 0, stream>>>(Wq_w, Wqh);
    cvt_f32_f16<<<(DM * DM / 4) / 256, 256, 0, stream>>>(Wk_w, Wkh);
    cvt_f32_f16<<<(DM * DM / 4) / 256, 256, 0, stream>>>(Wv_w, Wvh);
    cvt_f32_f16<<<(DM * DM / 4) / 256, 256, 0, stream>>>(Wo_w, Woh);

    // 2) Q/K/V projections (WMMA GEMM, async-LDS pipeline, f16 out)
    dim3 gg(MTOK / 128, DM / 128);
    gemm_xwT<true><<<gg, 256, 0, stream>>>(Xh, Wqh, Wq_b, Qh, nullptr, MTOK, DM, DM);
    gemm_xwT<true><<<gg, 256, 0, stream>>>(Xh, Wkh, Wk_b, Kh, nullptr, MTOK, DM, DM);
    gemm_xwT<true><<<gg, 256, 0, stream>>>(Xh, Wvh, Wv_b, Vh, nullptr, MTOK, DM, DM);

    // 3) gate projections, head-normalize, v*w
    wr_proj<<<65536, 256, 0, stream>>>(inp, Ww_w, Ww_b, Wr_w, Wr_b, wbf, rbf);
    norm_qk<<<65536, 256, 0, stream>>>(Qh, Kh);
    scale_v<<<(MTOK * DM) / 256, 256, 0, stream>>>(Vh, wbf);

    // 4) A accumulation (writes f32 A output + f16 copy)
    accum_A<<<B_SZ * NH, 256, 0, stream>>>(Kh, Vh, A_f, Ah);

    // 5) apply A, scale by r
    apply_A<<<dim3(B_SZ * NH, S_LEN / 128), 256, 0, stream>>>(Qh, Ah, rbf, Mid);

    // 6) final O projection (f32 out to d_out)
    gemm_xwT<false><<<gg, 256, 0, stream>>>(Mid, Woh, Wo_b, nullptr, out_f, MTOK, DM, DM);
}
